// simpleLSTM_20727512171210
// MI455X (gfx1250) — matrix operand order repaired and checked by reading, not compiled, not run on hardware
//
#include <hip/hip_runtime.h>

typedef __bf16 bf16_t;
typedef __attribute__((ext_vector_type(16))) __bf16 v16bf;
typedef __attribute__((ext_vector_type(8)))  __bf16 v8bf;
typedef __attribute__((ext_vector_type(8)))  float  v8f;
typedef __attribute__((ext_vector_type(4)))  unsigned int v4u;

#define B_   64
#define T_   1024
#define I_   512
#define H_   1024
#define K_   1536
#define KT_  48     // K_/32
#define KTX  16     // I_/32 (x part of K loop)
#define NCOLS 4096  // 4 gates * H
#define NBLOCKS 64
#define NTHREADS 128

#define XBF_ELEMS  (B_ * T_ * I_)   // 33554432
#define WT_ELEMS   (K_ * NCOLS)     // 6291456
#define XBF_BYTES  ((size_t)XBF_ELEMS * 2)
#define WT_BYTES   ((size_t)WT_ELEMS * 2)
#define HBUF_ELEMS (B_ * H_)
#define HBUF_BYTES ((size_t)HBUF_ELEMS * 2)

#define LDS_W_ELEMS (4 * KT_ * 512)            // 98304 bf16 = 192 KB
#define LDS_BYTES   ((size_t)LDS_W_ELEMS * 2)

// ---------------- prep: fp32 -> bf16 input conversion ----------------
__global__ void cvt_x_kernel(const float* __restrict__ x, bf16_t* __restrict__ xb, int n) {
    int i = (blockIdx.x * blockDim.x + threadIdx.x) * 8;
    if (i + 8 <= n) {
        float4 a = *(const float4*)(x + i);
        float4 b = *(const float4*)(x + i + 4);
        v8bf o;
        o[0] = (bf16_t)a.x; o[1] = (bf16_t)a.y; o[2] = (bf16_t)a.z; o[3] = (bf16_t)a.w;
        o[4] = (bf16_t)b.x; o[5] = (bf16_t)b.y; o[6] = (bf16_t)b.z; o[7] = (bf16_t)b.w;
        *(v8bf*)(xb + i) = o;
    }
}

// ---------------- prep: pack 4 weight matrices into WMMA B-tile layout ----------------
// Tiled layout: Wt[((kt*256 + cTile)*32 + lane)*16 + s], cTile = gate*64 + ct.
// Inside a tile: lane holds B[K][N=lane&15], K = 8*(lane>>4) + (s&7) + 16*(s>>3).
__global__ void pack_w_kernel(const float* __restrict__ Wf, const float* __restrict__ Wi,
                              const float* __restrict__ Wo, const float* __restrict__ Wc,
                              bf16_t* __restrict__ Wt) {
    int o = blockIdx.x * blockDim.x + threadIdx.x;   // < 6291456
    int n     = (o >> 4) & 15;
    int kr    = ((o >> 8) & 1) * 8 + (o & 7) + ((o >> 3) & 1) * 16;
    int cTile = (o >> 9) & 255;
    int kt    = o >> 17;                              // 0..47
    int k   = kt * 32 + kr;
    int col = cTile * 16 + n;                         // 0..4095
    int g   = col >> 10;
    int hc  = col & 1023;
    const float* W = (g == 0) ? Wf : (g == 1) ? Wi : (g == 2) ? Wo : Wc;
    Wt[o] = (bf16_t)W[(size_t)hc * K_ + k];
}

__global__ void init_sync_kernel(unsigned* s) { s[0] = 0u; s[1] = 0u; }

// ---------------- grid-wide barrier (generation counter) ----------------
__device__ __forceinline__ void grid_barrier(unsigned* cnt, unsigned* gen) {
    __syncthreads();
    if (threadIdx.x == 0) {
        __threadfence();
        unsigned g = atomicAdd(gen, 0u);
        if (atomicAdd(cnt, 1u) == NBLOCKS - 1u) {
            atomicExch(cnt, 0u);
            __threadfence();
            atomicAdd(gen, 1u);
        } else {
            while (atomicAdd(gen, 0u) == g) { __builtin_amdgcn_s_sleep(2); }
        }
        __threadfence();
    }
    __syncthreads();
}

__device__ __forceinline__ float sigmoidf_fast(float x) {
    return 1.0f / (1.0f + __expf(-x));
}
__device__ __forceinline__ float tanhf_fast(float x) {
    return 1.0f - 2.0f / (__expf(2.0f * x) + 1.0f);
}

__device__ __forceinline__ v16bf combine128(v4u lo, v4u hi) {
    union { v4u u[2]; v16bf v; } c;
    c.u[0] = lo; c.u[1] = hi;
    return c.v;
}

// Load the 4 gate B-tiles of one K-step straight from LDS.
// asm volatile: cannot be hoisted out of the t-loop (no LICM -> no spills),
// and guarantees the true DS path (ds_load_b128, DScnt) instead of flat/SYS.
// Tile stride = 1024B (512 bf16); per-lane row = 32B at addr (+16 for hi half).
__device__ __forceinline__ void lds_load_4tiles(unsigned addr, v16bf& b0, v16bf& b1,
                                                v16bf& b2, v16bf& b3) {
    v4u a0, a1, a2, a3, a4, a5, a6, a7;
    asm volatile(
        "ds_load_b128 %0, %8\n\t"
        "ds_load_b128 %1, %8 offset:16\n\t"
        "ds_load_b128 %2, %8 offset:1024\n\t"
        "ds_load_b128 %3, %8 offset:1040\n\t"
        "ds_load_b128 %4, %8 offset:2048\n\t"
        "ds_load_b128 %5, %8 offset:2064\n\t"
        "ds_load_b128 %6, %8 offset:3072\n\t"
        "ds_load_b128 %7, %8 offset:3088\n\t"
        "s_wait_dscnt 0x0"
        : "=v"(a0), "=v"(a1), "=v"(a2), "=v"(a3),
          "=v"(a4), "=v"(a5), "=v"(a6), "=v"(a7)
        : "v"(addr));
    b0 = combine128(a0, a1);
    b1 = combine128(a2, a3);
    b2 = combine128(a4, a5);
    b3 = combine128(a6, a7);
}

// ---------------- persistent LSTM kernel ----------------
// 64 blocks x 4 waves. Block = one 16-wide hidden-column tile (ct = blockIdx.x);
// wave-in-block = batch tile mt (rows mt*16..+15). The block's weight slice
// (4 gates x 48 K-tiles = 192KB bf16) lives in LDS for the entire sequence.
// Cell state c stays in registers for all 1024 steps.
__global__ void __launch_bounds__(NTHREADS, 1)
lstm_persistent_kernel(const bf16_t* __restrict__ xbf,
                       const bf16_t* __restrict__ Wt,
                       const float* __restrict__ bF_, const float* __restrict__ bI_,
                       const float* __restrict__ bO_, const float* __restrict__ bC_,
                       bf16_t* __restrict__ h0, bf16_t* __restrict__ h1,
                       unsigned* __restrict__ sync,
                       float* __restrict__ outH, float* __restrict__ outC) {
    extern __shared__ __align__(16) char lds_raw[];
    bf16_t* lWt = (bf16_t*)lds_raw;

    const int lane = threadIdx.x & 31;
    const int mt   = threadIdx.x >> 5;   // 0..3 batch tile
    const int ct   = blockIdx.x;         // 0..63 hidden-column tile
    const int m0   = mt * 16;
    const int half = lane >> 4;
    const int ln   = lane & 15;
    const int rA   = m0 + ln;            // A-matrix row this lane feeds
    const int cn   = ct * 16 + ln;       // output column (C/D layout: N = lane&15)
    const int koff = half * 8;           // ISA A layout: lane>=16 starts at K=8

    // ---- stage this block's weight slice into LDS (once) ----
    for (int c = threadIdx.x; c < LDS_W_ELEMS / 16; c += NTHREADS) {
        int tile = c >> 5;               // 0..191 = kt*4+g
        int l16  = c & 31;               // lane row inside tile
        int kt   = tile >> 2;
        int g    = tile & 3;
        const v16bf* src = (const v16bf*)(Wt + ((size_t)(kt * 256 + g * 64 + ct) * 512) + l16 * 16);
        *(v16bf*)(lWt + (size_t)tile * 512 + l16 * 16) = *src;
    }

    const float bF = bF_[cn], bI = bI_[cn], bO = bO_[cn], bC = bC_[cn];

    const bf16_t* xrow = xbf + (size_t)rA * T_ * I_;
    const size_t  hoff = (size_t)rA * H_;

    float creg[8];
#pragma unroll
    for (int j = 0; j < 8; ++j) creg[j] = 0.0f;

    // zero h_0 (buffer 0); blocks/waves jointly cover all of [64 x 1024]
#pragma unroll
    for (int j = 0; j < 8; ++j)
        h0[(size_t)(m0 + half * 8 + j) * H_ + cn] = (bf16_t)0.0f;

    __syncthreads();                     // LDS weights ready
    unsigned* cnt = sync;
    unsigned* gen = sync + 1;
    grid_barrier(cnt, gen);              // h_0 visible everywhere

    // 32-bit DS address: generic LDS pointer low bits = wave-relative LDS offset
    const unsigned ldsbase = (unsigned)(size_t)lWt + (unsigned)lane * 32u;

    for (int t = 0; t < T_; ++t) {
        const bf16_t* hprev = (t & 1) ? h1 : h0;
        bf16_t*       hnext = (t & 1) ? h0 : h1;

        v8f acc0 = {0.f,0.f,0.f,0.f,0.f,0.f,0.f,0.f};
        v8f acc1 = acc0, acc2 = acc0, acc3 = acc0;

        // ---- K loop, x part (K = 0..511) ----
        const bf16_t* xa = xrow + (size_t)t * I_;
        __builtin_prefetch(xa + I_, 0, 0);   // next timestep's x row
        for (int kt = 0; kt < KTX; ++kt) {
            const int k = kt * 32;
            v8bf alo = *(const v8bf*)(xa + k + koff);
            v8bf ahi = *(const v8bf*)(xa + k + koff + 16);
            v16bf a;
#pragma unroll
            for (int q = 0; q < 8; ++q) { a[q] = alo[q]; a[q + 8] = ahi[q]; }
            v16bf b0, b1, b2, b3;
            lds_load_4tiles(ldsbase + (unsigned)kt * 4096u, b0, b1, b2, b3);
            acc0 = __builtin_amdgcn_wmma_f32_16x16x32_bf16(false, a, false, b0, (short)0, acc0, false, false);
            acc1 = __builtin_amdgcn_wmma_f32_16x16x32_bf16(false, a, false, b1, (short)0, acc1, false, false);
            acc2 = __builtin_amdgcn_wmma_f32_16x16x32_bf16(false, a, false, b2, (short)0, acc2, false, false);
            acc3 = __builtin_amdgcn_wmma_f32_16x16x32_bf16(false, a, false, b3, (short)0, acc3, false, false);
        }

        // ---- K loop, h part (K = 512..1535) ----
        const bf16_t* ha = hprev + hoff;
        for (int kt = KTX; kt < KT_; ++kt) {
            const int kh = (kt - KTX) * 32;
            v8bf alo = *(const v8bf*)(ha + kh + koff);
            v8bf ahi = *(const v8bf*)(ha + kh + koff + 16);
            v16bf a;
#pragma unroll
            for (int q = 0; q < 8; ++q) { a[q] = alo[q]; a[q + 8] = ahi[q]; }
            v16bf b0, b1, b2, b3;
            lds_load_4tiles(ldsbase + (unsigned)kt * 4096u, b0, b1, b2, b3);
            acc0 = __builtin_amdgcn_wmma_f32_16x16x32_bf16(false, a, false, b0, (short)0, acc0, false, false);
            acc1 = __builtin_amdgcn_wmma_f32_16x16x32_bf16(false, a, false, b1, (short)0, acc1, false, false);
            acc2 = __builtin_amdgcn_wmma_f32_16x16x32_bf16(false, a, false, b2, (short)0, acc2, false, false);
            acc3 = __builtin_amdgcn_wmma_f32_16x16x32_bf16(false, a, false, b3, (short)0, acc3, false, false);
        }

        // ---- elementwise gates; C/D layout: VGPR j -> row m0 + half*8 + j ----
#pragma unroll
        for (int j = 0; j < 8; ++j) {
            float f = sigmoidf_fast(acc0[j] + bF);
            float i = sigmoidf_fast(acc1[j] + bI);
            float o = sigmoidf_fast(acc2[j] + bO);
            float g = tanhf_fast(acc3[j] + bC);
            float c = f * creg[j] + i * g;
            creg[j] = c;
            float h = o * tanhf_fast(c);
            const size_t idx = (size_t)(m0 + half * 8 + j) * H_ + cn;
            hnext[idx] = (bf16_t)h;
            if (t == T_ - 1) { outH[idx] = h; outC[idx] = c; }
        }

        grid_barrier(cnt, gen);
    }
}

// ---------------- launch ----------------
extern "C" void kernel_launch(void* const* d_in, const int* in_sizes, int n_in,
                              void* d_out, int out_size, void* d_ws, size_t ws_size,
                              hipStream_t stream) {
    const float* x  = (const float*)d_in[0];
    const float* Wf = (const float*)d_in[1];
    const float* bf = (const float*)d_in[2];
    const float* Wi = (const float*)d_in[3];
    const float* bi = (const float*)d_in[4];
    const float* Wo = (const float*)d_in[5];
    const float* bo = (const float*)d_in[6];
    const float* Wc = (const float*)d_in[7];
    const float* bc = (const float*)d_in[8];

    char* ws = (char*)d_ws;
    bf16_t* xbf = (bf16_t*)(ws);
    bf16_t* Wt  = (bf16_t*)(ws + XBF_BYTES);
    bf16_t* h0  = (bf16_t*)(ws + XBF_BYTES + WT_BYTES);
    bf16_t* h1  = h0 + HBUF_ELEMS;
    unsigned* sync = (unsigned*)(ws + XBF_BYTES + WT_BYTES + 2 * HBUF_BYTES);

    float* outH = (float*)d_out;
    float* outC = outH + HBUF_ELEMS;

    cvt_x_kernel<<<XBF_ELEMS / (8 * 256), 256, 0, stream>>>(x, xbf, XBF_ELEMS);
    pack_w_kernel<<<WT_ELEMS / 256, 256, 0, stream>>>(Wf, Wi, Wo, Wc, Wt);
    init_sync_kernel<<<1, 1, 0, stream>>>(sync);
    lstm_persistent_kernel<<<NBLOCKS, NTHREADS, LDS_BYTES, stream>>>(
        xbf, Wt, bf, bi, bo, bc, h0, h1, sync, outH, outC);
}
